// FractalSemanticNetwork_7988639170804
// MI455X (gfx1250) — matrix operand order repaired and checked by reading, not compiled, not run on hardware
//
#include <hip/hip_runtime.h>
#include <hip/hip_bf16.h>
#include <math.h>

// ---------------------------------------------------------------------------
// CDNA5 (gfx1250) wave32 implementation of FractalSemanticNetwork forward.
// f16 activations/weights in global memory; WMMA f32_16x16x32_f16 everywhere;
// GEMM operand staging via Tensor Data Mover (tensor_load_to_lds, TENSORcnt)
// with double buffering.
// ---------------------------------------------------------------------------

typedef __attribute__((ext_vector_type(16))) _Float16 v16h;
typedef __attribute__((ext_vector_type(8)))  float    v8f;
typedef __attribute__((ext_vector_type(4)))  unsigned int u32x4;
typedef __attribute__((ext_vector_type(8)))  unsigned int u32x8;

__device__ __forceinline__ v8f wmma16x16x32(v16h a, v16h b, v8f c) {
  return __builtin_amdgcn_wmma_f32_16x16x32_f16(false, a, false, b, (short)0, c,
                                                false, false);
}

// A fragment (16x32, row-major f16, row stride `ld` halves).  ISA 7.12.2.
__device__ __forceinline__ v16h load_a_frag(const _Float16* a, int ld) {
  int lane = threadIdx.x & 31;
  int m    = lane & 15;
  int koff = (lane < 16) ? 0 : 16;
  v16h r;
#pragma unroll
  for (int e = 0; e < 16; ++e) {
    r[e]   = a[m * ld + koff + e];
  }
  return r;
}

// B fragment (32x16) from N x K row-major tile: B[k][n] = W[n][k].
__device__ __forceinline__ v16h load_b_frag_nk(const _Float16* w, int ld) {
  int lane = threadIdx.x & 31;
  int n    = lane & 15;
  int kb   = (lane < 16) ? 0 : 16;
  v16h r;
#pragma unroll
  for (int e = 0; e < 16; ++e) r[e] = w[n * ld + kb + e];
  return r;
}

__device__ __forceinline__ float gelu_exact(float x) {
  return 0.5f * x * (1.0f + erff(x * 0.70710678118654752f));
}
__device__ __forceinline__ float sigmoidf_(float x) {
  return 1.0f / (1.0f + __expf(-x));
}

// ---------------------------------------------------------------------------
// TDM: issue a 2D f16 tile load (tile0 x tile1, row stride strideElems) from
// global memory into LDS.  D# packing per CDNA5 ISA ch.8 (group0/group1);
// groups 2/3 zero-filled (2D tensor).  Tracked by TENSORcnt.
// ---------------------------------------------------------------------------
__device__ __forceinline__ void tdm_load_2d_f16(unsigned lds_off,
                                                const _Float16* gptr,
                                                unsigned tdim0, unsigned tdim1,
                                                unsigned tile0, unsigned tile1,
                                                unsigned strideElems) {
  unsigned long long ga = (unsigned long long)(uintptr_t)gptr;
  u32x4 g0;
  g0[0] = 1u;                                   // count=1, user descriptor
  g0[1] = lds_off;                              // lds_addr (bytes)
  g0[2] = (unsigned)ga;                         // global_addr[31:0]
  g0[3] = (unsigned)((ga >> 32) & 0x1FFFFFFu)   // global_addr[56:32]
          | (2u << 30);                         // type=2 ("image")
  u32x8 g1;
  g1[0] = (1u << 16);                           // data_size=1 (2 bytes)
  g1[1] = (tdim0 & 0xFFFFu) << 16;              // tensor_dim0[15:0]
  g1[2] = ((tdim0 >> 16) & 0xFFFFu) | ((tdim1 & 0xFFFFu) << 16);
  g1[3] = ((tdim1 >> 16) & 0xFFFFu) | (tile0 << 16);   // tile_dim0
  g1[4] = tile1;                                // tile_dim1 (tile_dim2 = 0)
  g1[5] = strideElems;                          // tensor_dim0_stride[31:0]
  g1[6] = 0u;                                   // stride hi / dim1_stride lo
  g1[7] = 0u;
  u32x4 gz = {0u, 0u, 0u, 0u};
  asm volatile("tensor_load_to_lds %0, %1, %2, %3"
               :
               : "s"(g0), "s"(g1), "s"(gz), "s"(gz)
               : "memory");
}

// ---------------------------------------------------------------------------
// GEMM (full tiles only; M%128==0, N%128==0, K%32==0):
//   C[M,N](f16) = act(A[M,K](f16) @ W[N,K](f16)^T + bias[N](f32))
// 128x128 block tile, 8 waves (4 row-groups x 2 col-groups) -> 8 WMMAs/k-step.
// Operand staging: double-buffered TDM loads issued by wave 0.
// ---------------------------------------------------------------------------
__global__ __launch_bounds__(256) void gemm_tdm_kernel(
    const _Float16* __restrict__ A, const _Float16* __restrict__ W,
    const float* __restrict__ bias, _Float16* __restrict__ C,
    int M, int N, int K, int ldc, int act) {
  __shared__ _Float16 As[2][128][32];  // 2 x 8 KB
  __shared__ _Float16 Ws[2][128][32];  // 2 x 8 KB

  int tid  = threadIdx.x;
  int wid  = tid >> 5;
  int lane = tid & 31;
  int wr   = wid & 3;
  int wc   = wid >> 2;

  int mBase = blockIdx.y * 128;
  int nBase = blockIdx.x * 128;

  v8f acc[2][4];
#pragma unroll
  for (int i = 0; i < 2; ++i)
#pragma unroll
    for (int t = 0; t < 4; ++t) acc[i][t] = v8f{};

  // prologue: issue loads for k0 = 0 into buffer 0
  if (tid < 32) {
    tdm_load_2d_f16((unsigned)(uintptr_t)&As[0][0][0],
                    A + (size_t)mBase * K, (unsigned)K, (unsigned)(M - mBase),
                    32u, 128u, (unsigned)K);
    tdm_load_2d_f16((unsigned)(uintptr_t)&Ws[0][0][0],
                    W + (size_t)nBase * K, (unsigned)K, (unsigned)(N - nBase),
                    32u, 128u, (unsigned)K);
  }

  int cur = 0;
  for (int k0 = 0; k0 < K; k0 += 32, cur ^= 1) {
    bool hasNext = (k0 + 32 < K);
    if (tid < 32) {
      if (hasNext) {
        tdm_load_2d_f16((unsigned)(uintptr_t)&As[cur ^ 1][0][0],
                        A + (size_t)mBase * K + k0 + 32, (unsigned)(K - k0 - 32),
                        (unsigned)(M - mBase), 32u, 128u, (unsigned)K);
        tdm_load_2d_f16((unsigned)(uintptr_t)&Ws[cur ^ 1][0][0],
                        W + (size_t)nBase * K + k0 + 32, (unsigned)(K - k0 - 32),
                        (unsigned)(N - nBase), 32u, 128u, (unsigned)K);
        // in-order completion: <=2 outstanding => current pair has landed
        __builtin_amdgcn_s_wait_tensorcnt(2);
      } else {
        __builtin_amdgcn_s_wait_tensorcnt(0);
      }
    }
    __syncthreads();

    v16h af0 = load_a_frag(&As[cur][wr * 32][0], 32);
    v16h af1 = load_a_frag(&As[cur][wr * 32 + 16][0], 32);
#pragma unroll
    for (int t = 0; t < 4; ++t) {
      v16h bf = load_b_frag_nk(&Ws[cur][wc * 64 + t * 16][0], 32);
      acc[0][t] = wmma16x16x32(af0, bf, acc[0][t]);
      acc[1][t] = wmma16x16x32(af1, bf, acc[1][t]);
    }
    __syncthreads();  // reads done before wave 0 re-fills this buffer
  }

  int n16 = lane & 15;
  int mo  = (lane >> 4) * 8;
#pragma unroll
  for (int i = 0; i < 2; ++i) {
#pragma unroll
    for (int t = 0; t < 4; ++t) {
      int gn   = nBase + wc * 64 + t * 16 + n16;
      float bv = bias ? bias[gn] : 0.0f;
#pragma unroll
      for (int r = 0; r < 8; ++r) {
        int gm  = mBase + wr * 32 + i * 16 + mo + r;
        float v = acc[i][t][r] + bv;
        if (act == 1)      v = gelu_exact(v);
        else if (act == 2) v = sigmoidf_(v);
        C[(size_t)gm * ldc + gn] = (_Float16)v;
      }
    }
  }
}

// ---------------------------------------------------------------------------
// Edge GEMM for tiny shapes (head MLPs: M=32, N in {3,64,128,256}).
// Fully guarded staging; out_f32 selects f32 (d_out) or f16 output.
// ---------------------------------------------------------------------------
__global__ __launch_bounds__(256) void gemm_edge_kernel(
    const _Float16* __restrict__ A, const _Float16* __restrict__ W,
    const float* __restrict__ bias, void* __restrict__ C,
    int M, int N, int K, int ldc, int act, int out_f32) {
  __shared__ _Float16 As[128][32];
  __shared__ _Float16 Ws[128][32];

  int tid  = threadIdx.x;
  int wid  = tid >> 5;
  int lane = tid & 31;
  int wr   = wid & 3;
  int wc   = wid >> 2;
  int mBase = blockIdx.y * 128;
  int nBase = blockIdx.x * 128;

  v8f acc[2][4];
#pragma unroll
  for (int i = 0; i < 2; ++i)
#pragma unroll
    for (int t = 0; t < 4; ++t) acc[i][t] = v8f{};

  for (int k0 = 0; k0 < K; k0 += 32) {
    for (int i = tid; i < 128 * 32; i += 256) {
      int rr = i >> 5, cc = i & 31;
      int gm = mBase + rr, gk = k0 + cc;
      As[rr][cc] = (gm < M && gk < K) ? A[(size_t)gm * K + gk] : (_Float16)0.0f;
    }
    for (int i = tid; i < 128 * 32; i += 256) {
      int rr = i >> 5, cc = i & 31;
      int gn = nBase + rr, gk = k0 + cc;
      Ws[rr][cc] = (gn < N && gk < K) ? W[(size_t)gn * K + gk] : (_Float16)0.0f;
    }
    __syncthreads();
    v16h af0 = load_a_frag(&As[wr * 32][0], 32);
    v16h af1 = load_a_frag(&As[wr * 32 + 16][0], 32);
#pragma unroll
    for (int t = 0; t < 4; ++t) {
      v16h bf = load_b_frag_nk(&Ws[wc * 64 + t * 16][0], 32);
      acc[0][t] = wmma16x16x32(af0, bf, acc[0][t]);
      acc[1][t] = wmma16x16x32(af1, bf, acc[1][t]);
    }
    __syncthreads();
  }

  int n16 = lane & 15;
  int mo  = (lane >> 4) * 8;
#pragma unroll
  for (int i = 0; i < 2; ++i) {
#pragma unroll
    for (int t = 0; t < 4; ++t) {
      int gn = nBase + wc * 64 + t * 16 + n16;
      if (gn >= N) continue;
      float bv = bias ? bias[gn] : 0.0f;
#pragma unroll
      for (int r = 0; r < 8; ++r) {
        int gm = mBase + wr * 32 + i * 16 + mo + r;
        if (gm >= M) continue;
        float v = acc[i][t][r] + bv;
        if (act == 1)      v = gelu_exact(v);
        else if (act == 2) v = sigmoidf_(v);
        if (out_f32) ((float*)C)[(size_t)gm * ldc + gn] = v;
        else         ((_Float16*)C)[(size_t)gm * ldc + gn] = (_Float16)v;
      }
    }
  }
}

// ---------------------------------------------------------------------------
// Weight convert/pad: dst[r, 0..Kd) = (c < Ks) ? f16(src[r, c]) : 0
// ---------------------------------------------------------------------------
__global__ void wconv_kernel(const float* __restrict__ src,
                             _Float16* __restrict__ dst, int rows, int Ks,
                             int Kd) {
  int i = blockIdx.x * blockDim.x + threadIdx.x;
  if (i >= rows * Kd) return;
  int r = i / Kd, c = i % Kd;
  dst[i] = (c < Ks) ? (_Float16)src[(size_t)r * Ks + c] : (_Float16)0.0f;
}

// ---------------------------------------------------------------------------
// LayerNorm (f16 in/out, f32 math): out = LN(x (+res)) * g + b (* gate)
// ---------------------------------------------------------------------------
__global__ __launch_bounds__(256) void ln_kernel(
    const _Float16* __restrict__ x, const _Float16* __restrict__ res,
    const float* __restrict__ gamma, const float* __restrict__ beta,
    const _Float16* __restrict__ gate, _Float16* __restrict__ out,
    int ld_in, int ld_out) {
  __shared__ float s1[256], s2[256];
  int row = blockIdx.x;
  int e   = threadIdx.x;
  float v = (float)x[(size_t)row * ld_in + e];
  if (res) v += (float)res[(size_t)row * 256 + e];
  s1[e] = v;
  s2[e] = v * v;
  __syncthreads();
  for (int st = 128; st > 0; st >>= 1) {
    if (e < st) { s1[e] += s1[e + st]; s2[e] += s2[e + st]; }
    __syncthreads();
  }
  float mean = s1[0] * (1.0f / 256.0f);
  float var  = s2[0] * (1.0f / 256.0f) - mean * mean;
  float y    = (v - mean) * rsqrtf(var + 1e-5f) * gamma[e] + beta[e];
  if (gate) y *= (float)gate[(size_t)row * 256 + e];
  out[(size_t)row * ld_out + e] = (_Float16)y;
}

// ---------------------------------------------------------------------------
// feat(f16, K padded to 320) = concat(returns[128], volumes[128], frac[50], 0)
// ---------------------------------------------------------------------------
__global__ void concat_feat_kernel(const float* __restrict__ r,
                                   const float* __restrict__ vo,
                                   const float* __restrict__ f,
                                   _Float16* __restrict__ feat, int total) {
  int i = blockIdx.x * blockDim.x + threadIdx.x;
  if (i >= total) return;
  int t = i / 320, c = i % 320;
  float v = (c < 128)   ? r[(size_t)t * 128 + c]
            : (c < 256) ? vo[(size_t)t * 128 + (c - 128)]
            : (c < 306) ? f[(size_t)t * 50 + (c - 256)]
                        : 0.0f;
  feat[i] = (_Float16)v;
}

// ---------------------------------------------------------------------------
// Tiny NS=5 attention (4 heads, d=64), < 1 GFLOP total, scalar.
// ---------------------------------------------------------------------------
__global__ void ns_attn_kernel(const _Float16* __restrict__ qkv,
                               _Float16* __restrict__ o) {
  int idx = blockIdx.x * blockDim.x + threadIdx.x;  // t*20 + h*5 + q
  if (idx >= 16384 * 4 * 5) return;
  int q = idx % 5;
  int h = (idx / 5) & 3;
  int t = idx / 20;
  const _Float16* base = qkv + (size_t)(t * 5) * 768;
  const _Float16* qp   = base + q * 768 + h * 64;
  float s[5];
  float mx = -1e30f;
#pragma unroll
  for (int j = 0; j < 5; ++j) {
    const _Float16* kp = base + j * 768 + 256 + h * 64;
    float d = 0.0f;
    for (int e = 0; e < 64; ++e) d += (float)qp[e] * (float)kp[e];
    s[j] = d * 0.125f;
    mx   = fmaxf(mx, s[j]);
  }
  float sum = 0.0f;
#pragma unroll
  for (int j = 0; j < 5; ++j) { s[j] = __expf(s[j] - mx); sum += s[j]; }
  float inv = 1.0f / sum;
  _Float16* op = o + (size_t)(t * 5 + q) * 256 + h * 64;
  for (int e = 0; e < 64; ++e) {
    float acc = 0.0f;
#pragma unroll
    for (int j = 0; j < 5; ++j)
      acc += s[j] * (float)base[j * 768 + 512 + h * 64 + e];
    op[e] = (_Float16)(acc * inv);
  }
}

// ---------------------------------------------------------------------------
// Flash attention S=512, 8 heads, d=32; WMMA for QK^T and PV.
// Block = 4 waves = (batch, head, 64 q rows); wave = 16 q rows.
// K tile row-major, V tile transposed in LDS (contiguous fragment runs).
// ---------------------------------------------------------------------------
__global__ __launch_bounds__(128) void attn512_kernel(
    const _Float16* __restrict__ qkv, _Float16* __restrict__ o) {
  int qt = blockIdx.x & 7;
  int h  = (blockIdx.x >> 3) & 7;
  int b  = blockIdx.x >> 6;

  int tid   = threadIdx.x;
  int wid   = tid >> 5;
  int lane  = tid & 31;
  int n16   = lane & 15;
  int half  = lane >> 4;
  int kb    = half * 16;
  int qrow0 = b * 512 + qt * 64 + wid * 16;

  __shared__ _Float16 Ks[32][32];
  __shared__ _Float16 Vt[32][32];
  __shared__ _Float16 pbuf[4][16][32];

  int sr = tid >> 2;       // 0..31 key row
  int sc = (tid & 3) * 8;  // 0/8/16/24 dim col

  // Q fragment (16x32), pre-scaled by 1/sqrt(32)
  v16h qf;
  {
    const _Float16* qp = qkv + (size_t)(qrow0 + n16) * 768 + (size_t)h * 32;
    int koff = half ? 16 : 0;
#pragma unroll
    for (int e = 0; e < 16; ++e) {
      qf[e]  = (_Float16)((float)qp[koff + e] * 0.17677669529663689f);
    }
  }

  v8f o0 = {}, o1 = {};
  float run_m[8], run_l[8];
#pragma unroll
  for (int r = 0; r < 8; ++r) { run_m[r] = -1e30f; run_l[r] = 0.0f; }

  for (int j0 = 0; j0 < 512; j0 += 32) {
    __syncthreads();
    {
      const _Float16* kp =
          qkv + (size_t)(b * 512 + j0 + sr) * 768 + 256 + h * 32 + sc;
      const _Float16* vp =
          qkv + (size_t)(b * 512 + j0 + sr) * 768 + 512 + h * 32 + sc;
      *(uint4*)&Ks[sr][sc] = *(const uint4*)kp;  // 8 halves, 16B aligned
#pragma unroll
      for (int j = 0; j < 8; ++j) Vt[sc + j][sr] = vp[j];
    }
    __syncthreads();

    v16h kf0, kf1, vf0, vf1;
#pragma unroll
    for (int e = 0; e < 16; ++e) {
      kf0[e] = Ks[n16][kb + e];
      kf1[e] = Ks[16 + n16][kb + e];
      vf0[e] = Vt[n16][kb + e];
      vf1[e] = Vt[16 + n16][kb + e];
    }

    v8f zero = {};
    v8f s0 = wmma16x16x32(qf, kf0, zero);
    v8f s1 = wmma16x16x32(qf, kf1, zero);

#pragma unroll
    for (int r = 0; r < 8; ++r) {
      float tm = fmaxf(s0[r], s1[r]);
#pragma unroll
      for (int off = 1; off < 16; off <<= 1)
        tm = fmaxf(tm, __shfl_xor(tm, off, 32));
      float nm = fmaxf(run_m[r], tm);
      float al = __expf(run_m[r] - nm);
      run_m[r] = nm;
      float p0 = __expf(s0[r] - nm);
      float p1 = __expf(s1[r] - nm);
      float rs = p0 + p1;
#pragma unroll
      for (int off = 1; off < 16; off <<= 1) rs += __shfl_xor(rs, off, 32);
      run_l[r] = run_l[r] * al + rs;
      int m = r + (half ? 8 : 0);
      pbuf[wid][m][n16]      = (_Float16)p0;
      pbuf[wid][m][16 + n16] = (_Float16)p1;
      o0[r] *= al;
      o1[r] *= al;
    }
    __syncthreads();

    v16h pf = load_a_frag(&pbuf[wid][0][0], 32);
    o0 = wmma16x16x32(pf, vf0, o0);
    o1 = wmma16x16x32(pf, vf1, o1);
  }

#pragma unroll
  for (int r = 0; r < 8; ++r) {
    float inv = 1.0f / run_l[r];
    int m = r + (half ? 8 : 0);
    size_t orow = (size_t)(qrow0 + m) * 256 + (size_t)h * 32;
    o[orow + n16]      = (_Float16)(o0[r] * inv);
    o[orow + 16 + n16] = (_Float16)(o1[r] * inv);
  }
}

// ---------------------------------------------------------------------------
// final = x[:, -1, :] -> f16 scratch (head GEMM input) + f32 d_out tail
// ---------------------------------------------------------------------------
__global__ void gather_final_kernel(const _Float16* __restrict__ x,
                                    _Float16* __restrict__ fin,
                                    float* __restrict__ out_final) {
  int i = blockIdx.x * blockDim.x + threadIdx.x;
  if (i >= 32 * 256) return;
  int b = i >> 8, e = i & 255;
  float v = (float)x[((size_t)b * 512 + 511) * 256 + e];
  fin[i]       = (_Float16)v;
  out_final[i] = v;
}

// ---------------------------------------------------------------------------
// Host-side launch.  d_in leaf order:
//  0 returns 1 volumes 2 fractal | 3..32 scales (W1,b1,W2,b2,ln_g,ln_b)x5 |
//  33..36 attn | 37,38 gate | 39..42 proj | 43..90 layers x4 (12 each) |
//  91..94 ret | 95..98 reg | 99..102 hur
// ---------------------------------------------------------------------------
extern "C" void kernel_launch(void* const* d_in, const int* in_sizes, int n_in,
                              void* d_out, int out_size, void* d_ws,
                              size_t ws_size, hipStream_t stream) {
  (void)in_sizes; (void)n_in; (void)out_size; (void)ws_size;
  auto F = [&](int i) { return (const float*)d_in[i]; };
  float* out = (float*)d_out;
  const int T = 32 * 512;

  char* wsb  = (char*)d_ws;
  size_t off = 0;
  auto allocH = [&](size_t n) {
    void* p = wsb + off;
    off += (n * sizeof(_Float16) + 255) & ~(size_t)255;
    return (_Float16*)p;
  };

  // ---- f16 weights (converted every call; deterministic) ----
  _Float16* wsc[5][2];
  for (int s = 0; s < 5; ++s) {
    wsc[s][0] = allocH(256 * 320);
    wsc[s][1] = allocH(256 * 256);
  }
  _Float16* wqkv5 = allocH(768 * 256);
  _Float16* wo5   = allocH(256 * 256);
  _Float16* wgate = allocH((size_t)256 * 1280);
  _Float16* wproj = allocH((size_t)256 * 1280);
  _Float16* wL[4][4];
  for (int l = 0; l < 4; ++l) {
    wL[l][0] = allocH(768 * 256);            // Wqkv
    wL[l][1] = allocH(256 * 256);            // Wo
    wL[l][2] = allocH((size_t)1024 * 256);   // Wf1
    wL[l][3] = allocH((size_t)256 * 1024);   // Wf2
  }
  _Float16* whd[6];
  whd[0] = allocH(256 * 256);  // ret.W1
  whd[1] = allocH(128 * 256);  // ret.W2
  whd[2] = allocH(128 * 256);  // reg.W1
  whd[3] = allocH(3 * 128);    // reg.W2
  whd[4] = allocH(64 * 256);   // hur.W1
  whd[5] = allocH(128 * 64);   // hur.W2

  // ---- f16 activations ----
  _Float16* feat = allocH((size_t)T * 320);
  _Float16* z    = allocH((size_t)T * 1280);
  _Float16* q5   = allocH((size_t)T * 5 * 768);
  _Float16* a5   = allocH((size_t)T * 5 * 256);
  _Float16* flat = allocH((size_t)T * 1280);
  _Float16* g    = allocH((size_t)T * 256);
  _Float16* x    = allocH((size_t)T * 256);
  _Float16* t1   = allocH((size_t)T * 256);
  _Float16* t2   = allocH((size_t)T * 256);
  _Float16* ao   = allocH((size_t)T * 256);
  _Float16* big  = allocH((size_t)T * 1024);  // qkvL then ffh
  _Float16* fin  = allocH(32 * 256);
  _Float16* hh   = allocH(32 * 256);

  auto wc = [&](const float* src, _Float16* dst, int rows, int Ks, int Kd) {
    int total = rows * Kd;
    wconv_kernel<<<(total + 255) / 256, 256, 0, stream>>>(src, dst, rows, Ks, Kd);
  };
  // weight conversion
  for (int s = 0; s < 5; ++s) {
    wc(F(3 + s * 6 + 0), wsc[s][0], 256, 306, 320);
    wc(F(3 + s * 6 + 2), wsc[s][1], 256, 256, 256);
  }
  wc(F(33), wqkv5, 768, 256, 256);
  wc(F(35), wo5, 256, 256, 256);
  wc(F(37), wgate, 256, 1280, 1280);
  wc(F(39), wproj, 256, 1280, 1280);
  for (int l = 0; l < 4; ++l) {
    int p = 43 + l * 12;
    wc(F(p + 0), wL[l][0], 768, 256, 256);
    wc(F(p + 2), wL[l][1], 256, 256, 256);
    wc(F(p + 6), wL[l][2], 1024, 256, 256);
    wc(F(p + 8), wL[l][3], 256, 1024, 1024);
  }
  wc(F(91), whd[0], 256, 256, 256);
  wc(F(93), whd[1], 128, 256, 256);
  wc(F(95), whd[2], 128, 256, 256);
  wc(F(97), whd[3], 3, 128, 128);
  wc(F(99), whd[4], 64, 256, 256);
  wc(F(101), whd[5], 128, 64, 64);

  auto gemm = [&](const _Float16* A, const _Float16* W, const float* b,
                  _Float16* C, int M, int N, int K, int ldc, int act) {
    dim3 grid(N / 128, M / 128);
    gemm_tdm_kernel<<<grid, 256, 0, stream>>>(A, W, b, C, M, N, K, ldc, act);
  };
  auto gemm_edge = [&](const _Float16* A, const _Float16* W, const float* b,
                       void* C, int M, int N, int K, int ldc, int act,
                       int f32out) {
    dim3 grid((N + 127) / 128, (M + 127) / 128);
    gemm_edge_kernel<<<grid, 256, 0, stream>>>(A, W, b, C, M, N, K, ldc, act,
                                               f32out);
  };
  auto ln = [&](const _Float16* in_, const _Float16* res, const float* ga,
                const float* be, const _Float16* gt, _Float16* o_, int M,
                int ldi, int ldo) {
    ln_kernel<<<M, 256, 0, stream>>>(in_, res, ga, be, gt, o_, ldi, ldo);
  };

  // 1) feature concat (padded K=320)
  {
    int total = T * 320;
    concat_feat_kernel<<<(total + 255) / 256, 256, 0, stream>>>(
        F(0), F(1), F(2), feat, total);
  }

  // 2) 5 scale encoders -> z (T, 5, 256)
  for (int s = 0; s < 5; ++s) {
    int p = 3 + s * 6;
    gemm(feat, wsc[s][0], F(p + 1), t1, T, 256, 320, 256, 1);
    gemm(t1, wsc[s][1], F(p + 3), t2, T, 256, 256, 256, 0);
    ln(t2, nullptr, F(p + 4), F(p + 5), nullptr, z + s * 256, T, 256, 1280);
  }

  // 3) NS=5 MHA (4 heads, d=64)
  gemm(z, wqkv5, F(34), q5, T * 5, 768, 256, 768, 0);
  ns_attn_kernel<<<(T * 4 * 5 + 255) / 256, 256, 0, stream>>>(q5, a5);
  gemm(a5, wo5, F(36), flat, T * 5, 256, 256, 256, 0);  // flat = (T,1280)

  // 4) gate + projection
  gemm(flat, wgate, F(38), g, T, 256, 1280, 256, 2);
  gemm(flat, wproj, F(40), t1, T, 256, 1280, 256, 0);
  ln(t1, nullptr, F(41), F(42), g, x, T, 256, 256);

  // 5) 4 transformer layers
  for (int L = 0; L < 4; ++L) {
    int p = 43 + L * 12;
    gemm(x, wL[L][0], F(p + 1), big, T, 768, 256, 768, 0);
    attn512_kernel<<<32 * 8 * 8, 128, 0, stream>>>(big, ao);
    gemm(ao, wL[L][1], F(p + 3), t2, T, 256, 256, 256, 0);
    ln(t2, x, F(p + 4), F(p + 5), nullptr, x, T, 256, 256);
    gemm(x, wL[L][2], F(p + 7), big, T, 1024, 256, 1024, 1);
    gemm(big, wL[L][3], F(p + 9), t2, T, 256, 1024, 256, 0);
    ln(t2, x, F(p + 10), F(p + 11), nullptr, x, T, 256, 256);
  }

  // 6) final token + heads. d_out: ret[4096] | reg[96] | hur[4096] | final[8192]
  gather_final_kernel<<<32, 256, 0, stream>>>(x, fin, out + 4096 + 96 + 4096);

  gemm_edge(fin, whd[0], F(92), hh, 32, 256, 256, 256, 1, 0);
  gemm_edge(hh, whd[1], F(94), out, 32, 128, 256, 128, 0, 1);              // ret
  gemm_edge(fin, whd[2], F(96), hh, 32, 128, 256, 128, 1, 0);
  gemm_edge(hh, whd[3], F(98), out + 4096, 32, 3, 128, 3, 0, 1);           // reg
  gemm_edge(fin, whd[4], F(100), hh, 32, 64, 256, 64, 1, 0);
  gemm_edge(hh, whd[5], F(102), out + 4096 + 96, 32, 128, 64, 128, 2, 1);  // hur
}
